// AttentionModel_35510789603684
// MI455X (gfx1250) — compile-verified, matrix operand order checked by reading, not run on hardware
//
#include <hip/hip_runtime.h>
#include <hip/hip_bf16.h>

#define DEVINL __device__ __forceinline__

typedef __attribute__((ext_vector_type(16))) _Float16 v16h;
typedef __attribute__((ext_vector_type(8)))  float    v8f;

constexpr int Bn = 8;
constexpr int Tn = 2048;
constexpr int Fn = 257;
constexpr int Hn = 256;
constexpr int Rn = Bn * Tn;      // 16384 rows
constexpr int ATTN = 128;
constexpr int LDSW = 40;         // padded LDS row stride (halves); 80B keeps 16B align

// ---------------------------------------------------------------------------
// WMMA helpers (CDNA5 wave32, v_wmma_f32_16x16x32_f16)
// ---------------------------------------------------------------------------
DEVINL v8f wmma_16x16x32(v16h a, v16h b, v8f c) {
  return __builtin_amdgcn_wmma_f32_16x16x32_f16(
      /*neg_a=*/false, a, /*neg_b=*/false, b,
      /*c_mod=*/(short)0, c, /*reuse_a=*/false, /*reuse_b=*/false);
}

// A-fragment 16x32 f16 from row-major f16 matrix (ISA 7.12.2 layout):
// lane L: row = m0 + (L&15), kb = k0 + 8*(L>>4);
// elems 0..7 = A[row][kb+j], elems 8..15 = A[row][kb+16+j]
DEVINL v16h fragA_h(const _Float16* A, int lda, int m0, int k0) {
  int l = threadIdx.x & 31;
  const _Float16* p = A + (size_t)(m0 + (l & 15)) * lda + (k0 + ((l >> 4) << 3));
  v16h a;
#pragma unroll
  for (int j = 0; j < 8; ++j) { a[j] = p[j]; a[j + 8] = p[j + 16]; }
  return a;
}

DEVINL v16h fragA_h_clamp(const _Float16* A, int lda, int m0, int k0, int maxRow) {
  int l = threadIdx.x & 31;
  int row = m0 + (l & 15); if (row > maxRow) row = maxRow;
  const _Float16* p = A + (size_t)row * lda + (k0 + ((l >> 4) << 3));
  v16h a;
#pragma unroll
  for (int j = 0; j < 8; ++j) { a[j] = p[j]; a[j + 8] = p[j + 16]; }
  return a;
}

// stage a 64-row x 32-col f16 weight tile into LDS: 256 threads x 16B each.
// srcRow(i>>2) must give the global row; cols [k0, k0+32).
DEVINL void stage64x32(_Float16* Bs /* [64][LDSW] */, const _Float16* W, int ldw,
                       int rowBase, int k0) {
  int i = threadIdx.x;              // 0..255
  int row = i >> 2;                 // 0..63
  int chunk = i & 3;                // 0..3 (8 halves each)
  const int4* s = reinterpret_cast<const int4*>(W + (size_t)(rowBase + row) * ldw + k0 + chunk * 8);
  *reinterpret_cast<int4*>(Bs + row * LDSW + chunk * 8) = *s;
}

DEVINL float sigmoid_fast(float x) { return 1.0f / (1.0f + __expf(-x)); }
// tanh(x) = 1 - 2/(exp(2x)+1); saturates correctly at +-1
DEVINL float tanh_fast(float x) { return 1.0f - 2.0f / (__expf(2.0f * x) + 1.0f); }

// C layout: element r of v8f <-> M = r + 8*(lane>>4), N = lane&15
#define C_M(r) ((r) + (((threadIdx.x & 31) >> 4) << 3))
#define C_N    ((threadIdx.x & 31) & 15)

// ---------------------------------------------------------------------------
// utility: zero fill / f32 -> f16 convert
// ---------------------------------------------------------------------------
__global__ void k_zero_u32(unsigned* p, long n) {
  long i = (long)blockIdx.x * blockDim.x + threadIdx.x;
  long st = (long)gridDim.x * blockDim.x;
  for (; i < n; i += st) p[i] = 0u;
}

__global__ void k_cvt_h(const float* __restrict__ src, _Float16* __restrict__ dst, long n) {
  long i = (long)blockIdx.x * blockDim.x + threadIdx.x;
  long st = (long)gridDim.x * blockDim.x;
  for (; i < n; i += st) dst[i] = (_Float16)src[i];
}

// ---------------------------------------------------------------------------
// h = tanh(x @ feat_w.T + feat_b)   (16384 x 257)@(257 x 256)
// WG: 128 rows x 64 cols; weight tile staged in LDS, shared by 8 waves.
// ---------------------------------------------------------------------------
__global__ void k_feat(const _Float16* __restrict__ xh, const _Float16* __restrict__ Wfh,
                       const float* __restrict__ x, const float* __restrict__ Wf,
                       const float* __restrict__ bf, _Float16* __restrict__ hout) {
  __shared__ _Float16 Bs[64 * LDSW];
  int mb = (blockIdx.x >> 2) * 128;          // 128 m-groups
  int nb = (blockIdx.x & 3) * 64;            // 4 n-groups
  int w = threadIdx.x >> 5;
  int m0 = mb + w * 16;
  v8f acc[4] = {};
  for (int k0 = 0; k0 < 256; k0 += 32) {
    __syncthreads();
    stage64x32(Bs, Wfh, Fn, nb, k0);
    __syncthreads();
    v16h a = fragA_h(xh, Fn, m0, k0);
#pragma unroll
    for (int t = 0; t < 4; ++t) {
      v16h b = fragA_h(Bs, LDSW, t * 16, 0);
      acc[t] = wmma_16x16x32(a, b, acc[t]);
    }
  }
  // K remainder: k = 256 (scalar rank-1 update, f32 source)
  float xc[8];
#pragma unroll
  for (int r = 0; r < 8; ++r) xc[r] = x[(size_t)(m0 + C_M(r)) * Fn + 256];
#pragma unroll
  for (int t = 0; t < 4; ++t) {
    int n = nb + t * 16 + C_N;
    float w256 = Wf[(size_t)n * Fn + 256];
    float bias = bf[n];
#pragma unroll
    for (int r = 0; r < 8; ++r) {
      int m = m0 + C_M(r);
      hout[(size_t)m * Hn + n] = (_Float16)tanh_fast(acc[t][r] + xc[r] * w256 + bias);
    }
  }
}

// ---------------------------------------------------------------------------
// One LSTM scan step: gates = hx@Wih.T + hprev@Whh.T + b (xp fused)
// WG: 128 rows x 16 hidden units (4 gate tiles staged in LDS per K-step).
// ---------------------------------------------------------------------------
__global__ void k_lstm_step(const _Float16* __restrict__ hx,    // Tn x Hn
                            const _Float16* __restrict__ hprev, // Tn x Hn
                            float* __restrict__ cstate,          // Tn x Hn
                            const _Float16* __restrict__ Wih,    // 4H x H (f16)
                            const _Float16* __restrict__ Whh,    // 4H x H (f16)
                            const float* __restrict__ bih,
                            const float* __restrict__ bhh,
                            _Float16* __restrict__ hout) {       // Tn x Hn
  __shared__ _Float16 Bs[64 * LDSW];
  int mb = (blockIdx.x >> 4) * 128;          // 16 row groups
  int n0 = (blockIdx.x & 15) * 16;           // 16 hidden tiles
  int w = threadIdx.x >> 5;
  int m0 = mb + w * 16;
  v8f acc[4] = {};
  __builtin_prefetch(Wih + (size_t)n0 * Hn, 0, 1);
  // staged rows: gate g occupies LDS rows [g*16, g*16+16) = Wih[g*Hn + n0 + r]
  // stage64x32 rowBase trick: rows are NOT contiguous across gates in W, so
  // stage each gate's 16x32 sub-tile: thread i: row i>>2 -> gate (row>>4), r (row&15)
  for (int k0 = 0; k0 < Hn; k0 += 32) {
    __syncthreads();
    {
      int i = threadIdx.x, row = i >> 2, chunk = i & 3;
      int g = row >> 4, rr = row & 15;
      const int4* s = reinterpret_cast<const int4*>(
          Wih + (size_t)(g * Hn + n0 + rr) * Hn + k0 + chunk * 8);
      *reinterpret_cast<int4*>(Bs + row * LDSW + chunk * 8) = *s;
    }
    __syncthreads();
    v16h a = fragA_h(hx, Hn, m0, k0);
#pragma unroll
    for (int g = 0; g < 4; ++g) {
      v16h b = fragA_h(Bs, LDSW, g * 16, 0);
      acc[g] = wmma_16x16x32(a, b, acc[g]);
    }
  }
  for (int k0 = 0; k0 < Hn; k0 += 32) {
    __syncthreads();
    {
      int i = threadIdx.x, row = i >> 2, chunk = i & 3;
      int g = row >> 4, rr = row & 15;
      const int4* s = reinterpret_cast<const int4*>(
          Whh + (size_t)(g * Hn + n0 + rr) * Hn + k0 + chunk * 8);
      *reinterpret_cast<int4*>(Bs + row * LDSW + chunk * 8) = *s;
    }
    __syncthreads();
    v16h a = fragA_h(hprev, Hn, m0, k0);
#pragma unroll
    for (int g = 0; g < 4; ++g) {
      v16h b = fragA_h(Bs, LDSW, g * 16, 0);
      acc[g] = wmma_16x16x32(a, b, acc[g]);
    }
  }
  int n = n0 + C_N;
  float bs0 = bih[0 * Hn + n] + bhh[0 * Hn + n];
  float bs1 = bih[1 * Hn + n] + bhh[1 * Hn + n];
  float bs2 = bih[2 * Hn + n] + bhh[2 * Hn + n];
  float bs3 = bih[3 * Hn + n] + bhh[3 * Hn + n];
#pragma unroll
  for (int r = 0; r < 8; ++r) {
    int t = m0 + C_M(r);
    float gi = sigmoid_fast(acc[0][r] + bs0);
    float gf = sigmoid_fast(acc[1][r] + bs1);
    float gg = tanh_fast(acc[2][r] + bs2);
    float go = sigmoid_fast(acc[3][r] + bs3);
    size_t idx = (size_t)t * Hn + n;
    float c = gf * cstate[idx] + gi * gg;
    cstate[idx] = c;
    hout[idx] = (_Float16)(go * tanh_fast(c));
  }
}

// ---------------------------------------------------------------------------
// qs = q @ score_w.T   (16384 x 256)@(256 x 256); WG: 128 x 64 staged
// ---------------------------------------------------------------------------
__global__ void k_qs(const _Float16* __restrict__ q, const _Float16* __restrict__ Ws,
                     _Float16* __restrict__ qs) {
  __shared__ _Float16 Bs[64 * LDSW];
  int mb = (blockIdx.x >> 2) * 128;
  int nb = (blockIdx.x & 3) * 64;
  int w = threadIdx.x >> 5;
  int m0 = mb + w * 16;
  v8f acc[4] = {};
  for (int k0 = 0; k0 < Hn; k0 += 32) {
    __syncthreads();
    stage64x32(Bs, Ws, Hn, nb, k0);
    __syncthreads();
    v16h a = fragA_h(q, Hn, m0, k0);
#pragma unroll
    for (int t = 0; t < 4; ++t) {
      v16h b = fragA_h(Bs, LDSW, t * 16, 0);
      acc[t] = wmma_16x16x32(a, b, acc[t]);
    }
  }
#pragma unroll
  for (int t = 0; t < 4; ++t) {
    int n = nb + t * 16 + C_N;
#pragma unroll
    for (int r = 0; r < 8; ++r)
      qs[(size_t)(m0 + C_M(r)) * Hn + n] = (_Float16)acc[t][r];
  }
}

// ---------------------------------------------------------------------------
// rowmax over the FULL score row. 8 waves split 128 col tiles; A fragments of
// the row block hoisted into VGPRs once and reused for all tiles.
// ---------------------------------------------------------------------------
__global__ void k_rowmax(const _Float16* __restrict__ qs, const _Float16* __restrict__ kh,
                         float* __restrict__ rowmax) {
  int blk = blockIdx.x;          // 0..1023
  int bb = blk >> 7;
  int trb = blk & 127;
  int m0g = bb * Tn + trb * 16;
  int w = threadIdx.x >> 5;
  v16h afr[8];
#pragma unroll
  for (int kk = 0; kk < 8; ++kk) afr[kk] = fragA_h(qs, Hn, m0g, kk * 32);
  v8f vmax;
#pragma unroll
  for (int r = 0; r < 8; ++r) vmax[r] = -3.4e38f;
  for (int ct = w; ct < Tn / 16; ct += 8) {
    v8f acc = {};
    int n0g = bb * Tn + ct * 16;
#pragma unroll
    for (int kk = 0; kk < 8; ++kk) {
      v16h b = fragA_h(kh, Hn, n0g, kk * 32);  // s[t][s'] = sum_h qs[t][h]*k[s'][h]
      acc = wmma_16x16x32(afr[kk], b, acc);
    }
#pragma unroll
    for (int r = 0; r < 8; ++r) vmax[r] = fmaxf(vmax[r], acc[r]);
  }
#pragma unroll
  for (int off = 1; off <= 8; off <<= 1)
#pragma unroll
    for (int r = 0; r < 8; ++r)
      vmax[r] = fmaxf(vmax[r], __shfl_xor(vmax[r], off, 32));
  __shared__ float smax[8][16];
  int l = threadIdx.x & 31;
  if (l == 0)
#pragma unroll
    for (int r = 0; r < 8; ++r) smax[w][r] = vmax[r];
  if (l == 16)
#pragma unroll
    for (int r = 0; r < 8; ++r) smax[w][8 + r] = vmax[r];
  __syncthreads();
  if (threadIdx.x < 16) {
    float m = smax[0][threadIdx.x];
#pragma unroll
    for (int ww = 1; ww < 8; ++ww) m = fmaxf(m, smax[ww][threadIdx.x]);
    rowmax[m0g + threadIdx.x] = m;
  }
}

// ---------------------------------------------------------------------------
// Banded softmax + context: one block (4 waves) per 16-row block.
// ---------------------------------------------------------------------------
__global__ void k_band(const _Float16* __restrict__ qs, const _Float16* __restrict__ kh,
                       const float* __restrict__ rowmax,
                       float* __restrict__ wout,      // (B,T,T) slot of d_out
                       _Float16* __restrict__ catt) { // R x H
  int blk = blockIdx.x;
  int bb = blk >> 7;
  int trb = blk & 127;
  int t0 = trb * 16;
  int m0g = bb * Tn + t0;
  int s_lo = t0 - ATTN; if (s_lo < 0) s_lo = 0;
  int ct0 = s_lo >> 4;
  int cnt = trb - ct0 + 1;       // <= 9 column tiles in band

  __shared__ float    eLDS[16][160];
  __shared__ _Float16 wLDS[16][160];
  __shared__ float    rinv[16];
  for (int i = threadIdx.x; i < 16 * 160; i += blockDim.x) {
    eLDS[i / 160][i % 160] = 0.0f;
    wLDS[i / 160][i % 160] = (_Float16)0.0f;
  }
  __syncthreads();

  int w = threadIdx.x >> 5;
  int l = threadIdx.x & 31;

  v16h afr[8];
#pragma unroll
  for (int kk = 0; kk < 8; ++kk) afr[kk] = fragA_h(qs, Hn, m0g, kk * 32);
  float rm[8];
#pragma unroll
  for (int r = 0; r < 8; ++r) rm[r] = rowmax[m0g + C_M(r)];

  // phase 1: in-band score tiles -> e = exp(s - rowmax) * band
  for (int ci = w; ci < cnt; ci += 4) {
    int ct = ct0 + ci;
    int n0g = bb * Tn + ct * 16;
    v8f acc = {};
#pragma unroll
    for (int kk = 0; kk < 8; ++kk) {
      v16h b = fragA_h(kh, Hn, n0g, kk * 32);
      acc = wmma_16x16x32(afr[kk], b, acc);
    }
    int n = l & 15;
    int scol = ct * 16 + n;
#pragma unroll
    for (int r = 0; r < 8; ++r) {
      int t = t0 + C_M(r);
      float e = 0.0f;
      if (scol <= t && scol >= t - ATTN) e = __expf(acc[r] - rm[r]);
      eLDS[C_M(r)][ci * 16 + n] = e;
    }
  }
  __syncthreads();

  // phase 2: row sums
  if (threadIdx.x < 16) {
    float s = 0.0f;
    int cw = cnt * 16;
    for (int j = 0; j < cw; ++j) s += eLDS[threadIdx.x][j];
    rinv[threadIdx.x] = 1.0f / (s + 1e-10f);
  }
  __syncthreads();

  // phase 3: normalize; write w_ to global + f16 copy for the context GEMM
  {
    int cw = cnt * 16;
    for (int i = threadIdx.x; i < 16 * cw; i += blockDim.x) {
      int m = i / cw, j = i - m * cw;
      float wv = eLDS[m][j] * rinv[m];
      wLDS[m][j] = (_Float16)wv;
      long t = m0g + m;
      long scol = (long)ct0 * 16 + j;
      wout[t * (long)Tn + scol] = wv;
    }
  }
  __syncthreads();

  // phase 4: c(16x256) = wLDS(16 x Kp) @ k_rows(Kp x 256)
  int Kp = ((cnt * 16 + 31) >> 5) << 5;
  for (int ht = w; ht < Hn / 16; ht += 4) {
    int n0 = ht * 16;
    v8f acc = {};
    for (int k0 = 0; k0 < Kp; k0 += 32) {
      v16h a = fragA_h(&wLDS[0][0], 160, 0, k0);
      v16h b;   // true row-major B: B[k][n] = kh[bb*Tn + ct0*16 + k][n]
      {
        int n = n0 + (l & 15);
        int kb = k0 + ((l >> 4) << 3);
#pragma unroll
        for (int j = 0; j < 8; ++j) {
          int r1 = ct0 * 16 + kb + j;       if (r1 > Tn - 1) r1 = Tn - 1;
          int r2 = ct0 * 16 + kb + 16 + j;  if (r2 > Tn - 1) r2 = Tn - 1;
          b[j]     = kh[(size_t)(bb * Tn + r1) * Hn + n];
          b[j + 8] = kh[(size_t)(bb * Tn + r2) * Hn + n];
        }
      }
      acc = wmma_16x16x32(a, b, acc);
    }
    int n = n0 + C_N;
#pragma unroll
    for (int r = 0; r < 8; ++r)
      catt[(size_t)(m0g + C_M(r)) * Hn + n] = (_Float16)acc[r];
  }
}

// ---------------------------------------------------------------------------
// out_e = tanh(concat([c, q]) @ enhance_w.T + b)  K=512; WG: 128 x 64 staged
// ---------------------------------------------------------------------------
__global__ void k_enh(const _Float16* __restrict__ catt, const _Float16* __restrict__ q,
                      const _Float16* __restrict__ We, const float* __restrict__ be,
                      _Float16* __restrict__ oe) {
  __shared__ _Float16 Bs[64 * LDSW];
  int mb = (blockIdx.x >> 2) * 128;
  int nb = (blockIdx.x & 3) * 64;
  int w = threadIdx.x >> 5;
  int m0 = mb + w * 16;
  v8f acc[4] = {};
  for (int k0 = 0; k0 < Hn; k0 += 32) {
    __syncthreads();
    stage64x32(Bs, We, 2 * Hn, nb, k0);
    __syncthreads();
    v16h a = fragA_h(catt, Hn, m0, k0);
#pragma unroll
    for (int t = 0; t < 4; ++t) {
      v16h b = fragA_h(Bs, LDSW, t * 16, 0);
      acc[t] = wmma_16x16x32(a, b, acc[t]);
    }
  }
  for (int k0 = 0; k0 < Hn; k0 += 32) {
    __syncthreads();
    stage64x32(Bs, We, 2 * Hn, nb, Hn + k0);
    __syncthreads();
    v16h a = fragA_h(q, Hn, m0, k0);
#pragma unroll
    for (int t = 0; t < 4; ++t) {
      v16h b = fragA_h(Bs, LDSW, t * 16, 0);
      acc[t] = wmma_16x16x32(a, b, acc[t]);
    }
  }
#pragma unroll
  for (int t = 0; t < 4; ++t) {
    int n = nb + t * 16 + C_N;
    float bias = be[n];
#pragma unroll
    for (int r = 0; r < 8; ++r)
      oe[(size_t)(m0 + C_M(r)) * Hn + n] = (_Float16)tanh_fast(acc[t][r] + bias);
  }
}

// ---------------------------------------------------------------------------
// m = sigmoid(oe @ mask_w.T + mask_b); out0 = x * m     N = 257 (17 tiles)
// ---------------------------------------------------------------------------
__global__ void k_mask(const _Float16* __restrict__ oe, const _Float16* __restrict__ Wm,
                       const float* __restrict__ bm, const float* __restrict__ x,
                       float* __restrict__ out0) {
  int wave = (blockIdx.x * blockDim.x + threadIdx.x) >> 5;
  int mt = wave / 17;
  int nt = wave - mt * 17;
  int m0 = mt * 16;
  int n0 = nt * 16;
  v8f acc = {};
  for (int k0 = 0; k0 < Hn; k0 += 32) {
    v16h a = fragA_h(oe, Hn, m0, k0);
    v16h b = fragA_h_clamp(Wm, Hn, n0, k0, Fn - 1);   // clamp OOB rows of (257 x 256)
    acc = wmma_16x16x32(a, b, acc);
  }
  int n = n0 + C_N;
  if (n < Fn) {
    float bias = bm[n];
#pragma unroll
    for (int r = 0; r < 8; ++r) {
      int m = m0 + C_M(r);
      float mv = sigmoid_fast(acc[r] + bias);
      out0[(size_t)m * Fn + n] = x[(size_t)m * Fn + n] * mv;
    }
  }
}

// ---------------------------------------------------------------------------
extern "C" void kernel_launch(void* const* d_in, const int* in_sizes, int n_in,
                              void* d_out, int out_size, void* d_ws, size_t ws_size,
                              hipStream_t stream) {
  (void)in_sizes; (void)n_in; (void)out_size; (void)ws_size;
  const float* x      = (const float*)d_in[0];
  const float* feat_w = (const float*)d_in[1];
  const float* feat_b = (const float*)d_in[2];
  const float* kWih   = (const float*)d_in[3];
  const float* kWhh   = (const float*)d_in[4];
  const float* kbih   = (const float*)d_in[5];
  const float* kbhh   = (const float*)d_in[6];
  const float* qWih   = (const float*)d_in[7];
  const float* qWhh   = (const float*)d_in[8];
  const float* qbih   = (const float*)d_in[9];
  const float* qbhh   = (const float*)d_in[10];
  const float* Wsc    = (const float*)d_in[11];
  const float* Wen    = (const float*)d_in[12];
  const float* ben    = (const float*)d_in[13];
  const float* Wmk    = (const float*)d_in[14];
  const float* bmk    = (const float*)d_in[15];

  float* out0 = (float*)d_out;
  float* wout = out0 + (size_t)Rn * Fn;     // (B,T,T) attention weights

  // workspace carve (256B aligned)
  char* ws = (char*)d_ws;
  size_t off = 0;
  auto take = [&](size_t bytes) { char* p = ws + off; off += (bytes + 255) & ~(size_t)255; return p; };
  _Float16* h_h   = (_Float16*)take((size_t)Rn * Hn * 2);
  _Float16* k_h   = (_Float16*)take((size_t)Rn * Hn * 2);
  _Float16* q_h   = (_Float16*)take((size_t)Rn * Hn * 2);
  _Float16* qs_h  = (_Float16*)take((size_t)Rn * Hn * 2);
  _Float16* ca_h  = (_Float16*)take((size_t)Rn * Hn * 2);
  _Float16* oe_h  = (_Float16*)take((size_t)Rn * Hn * 2);
  _Float16* hz    = (_Float16*)take((size_t)Tn * Hn * 2);   // zero h0
  float*    c_k   = (float*)take((size_t)Tn * Hn * 4);
  float*    c_q   = (float*)take((size_t)Tn * Hn * 4);
  float*    rmax  = (float*)take((size_t)Rn * 4);
  _Float16* x_h   = (_Float16*)take((size_t)Rn * Fn * 2);   // f16 activations/weights
  _Float16* Wf_h  = (_Float16*)take((size_t)Hn * Fn * 2);
  _Float16* kWih_h= (_Float16*)take((size_t)4 * Hn * Hn * 2);
  _Float16* kWhh_h= (_Float16*)take((size_t)4 * Hn * Hn * 2);
  _Float16* qWih_h= (_Float16*)take((size_t)4 * Hn * Hn * 2);
  _Float16* qWhh_h= (_Float16*)take((size_t)4 * Hn * Hn * 2);
  _Float16* Wsc_h = (_Float16*)take((size_t)Hn * Hn * 2);
  _Float16* Wen_h = (_Float16*)take((size_t)Hn * 2 * Hn * 2);
  _Float16* Wmk_h = (_Float16*)take((size_t)Fn * Hn * 2);

  // zero: w_ output slot, LSTM cell states, h0
  k_zero_u32<<<2048, 256, 0, stream>>>((unsigned*)wout, (long)Rn * Tn);
  k_zero_u32<<<256, 256, 0, stream>>>((unsigned*)c_k, (long)Tn * Hn);
  k_zero_u32<<<256, 256, 0, stream>>>((unsigned*)c_q, (long)Tn * Hn);
  k_zero_u32<<<256, 256, 0, stream>>>((unsigned*)hz, (long)Tn * Hn / 2);

  // one-time f32 -> f16 conversions
  k_cvt_h<<<512, 256, 0, stream>>>(x,      x_h,    (long)Rn * Fn);
  k_cvt_h<<<64,  256, 0, stream>>>(feat_w, Wf_h,   (long)Hn * Fn);
  k_cvt_h<<<128, 256, 0, stream>>>(kWih,   kWih_h, (long)4 * Hn * Hn);
  k_cvt_h<<<128, 256, 0, stream>>>(kWhh,   kWhh_h, (long)4 * Hn * Hn);
  k_cvt_h<<<128, 256, 0, stream>>>(qWih,   qWih_h, (long)4 * Hn * Hn);
  k_cvt_h<<<128, 256, 0, stream>>>(qWhh,   qWhh_h, (long)4 * Hn * Hn);
  k_cvt_h<<<64,  256, 0, stream>>>(Wsc,    Wsc_h,  (long)Hn * Hn);
  k_cvt_h<<<64,  256, 0, stream>>>(Wen,    Wen_h,  (long)Hn * 2 * Hn);
  k_cvt_h<<<64,  256, 0, stream>>>(Wmk,    Wmk_h,  (long)Fn * Hn);

  // feature projection: 128 row-groups x 4 col-groups
  k_feat<<<512, 256, 0, stream>>>(x_h, Wf_h, x, feat_w, feat_b, h_h);

  // LSTM scans: 8 steps over axis 0 (batch = Tn per step)
  for (int s = 0; s < Bn; ++s) {
    const _Float16* hx = h_h + (size_t)s * Tn * Hn;
    const _Float16* kp = s ? k_h + (size_t)(s - 1) * Tn * Hn : hz;
    const _Float16* qp = s ? q_h + (size_t)(s - 1) * Tn * Hn : hz;
    k_lstm_step<<<256, 256, 0, stream>>>(hx, kp, c_k, kWih_h, kWhh_h, kbih, kbhh,
                                         k_h + (size_t)s * Tn * Hn);
    k_lstm_step<<<256, 256, 0, stream>>>(hx, qp, c_q, qWih_h, qWhh_h, qbih, qbhh,
                                         q_h + (size_t)s * Tn * Hn);
  }

  // qs = q @ score_w.T
  k_qs<<<512, 256, 0, stream>>>(q_h, Wsc_h, qs_h);

  // full-row max of scores (B*T/16 = 1024 row blocks)
  k_rowmax<<<1024, 256, 0, stream>>>(qs_h, k_h, rmax);

  // banded softmax + context
  k_band<<<1024, 128, 0, stream>>>(qs_h, k_h, rmax, wout, ca_h);

  // enhance + mask + final modulation
  k_enh<<<512, 256, 0, stream>>>(ca_h, q_h, Wen_h, ben, oe_h);
  k_mask<<<2176, 256, 0, stream>>>(oe_h, Wmk_h, bmk, x, out0);
}
